// BatchGCNLayer_20658792694236
// MI455X (gfx1250) — matrix operand order repaired and checked by reading, not compiled, not run on hardware
//
#include <hip/hip_runtime.h>

#define B_DIM 64
#define C_DIM 64
#define N_DIM 1024
#define O_DIM 128
#define KNN   49
#define BN_EPS 1e-5f

typedef __attribute__((ext_vector_type(16))) _Float16 v16h;
typedef __attribute__((ext_vector_type(8)))  _Float16 v8h;
typedef __attribute__((ext_vector_type(2)))  _Float16 v2h;
typedef __attribute__((ext_vector_type(8)))  float    v8f;

union AFrag { v16h v; v8h h[2]; };
union CFrag { v8f  v; float f[8]; };

__device__ __forceinline__ v8f wmma_f16(v16h a, v16h b, v8f c) {
  // D = A*B + C, fp32 accum. 8-arg form: (neg_a, A, neg_b, B, c_mod, C, reuse_a, reuse_b)
  return __builtin_amdgcn_wmma_f32_16x16x32_f16(false, a, false, b, (short)0, c, false, false);
}

__device__ __forceinline__ unsigned umin2(unsigned a, unsigned b) { return a < b ? a : b; }

// ---------------------------------------------------------------------------
// Kernel 0a: transpose X (B,C,N) fp32 -> Xh (B,N,C) f16, and squared norms.
// ---------------------------------------------------------------------------
__global__ void prep_nodes(const float* __restrict__ X,
                           _Float16* __restrict__ Xh,
                           float* __restrict__ sq) {
  const int tid = blockIdx.x * blockDim.x + threadIdx.x;
  if (tid >= B_DIM * N_DIM) return;
  const int b = tid >> 10;
  const int n = tid & (N_DIM - 1);
  const float* xp = X + (size_t)b * C_DIM * N_DIM + n;
  _Float16* op = Xh + (size_t)tid * C_DIM;
  float s = 0.f;
#pragma unroll 8
  for (int c = 0; c < C_DIM; ++c) {
    const float v = xp[(size_t)c * N_DIM];
    s += v * v;
    op[c] = (_Float16)v;
  }
  sq[tid] = s;
}

// ---------------------------------------------------------------------------
// Kernel 0b: convert W to f16 and zero the output accumulator.
// ---------------------------------------------------------------------------
__global__ void prep_small(const float* __restrict__ W,
                           _Float16* __restrict__ Wh,
                           float* __restrict__ out) {
  const int tid = blockIdx.x * blockDim.x + threadIdx.x;
  if (tid < O_DIM * C_DIM) Wh[tid] = (_Float16)W[tid];
  if (tid < B_DIM * O_DIM) out[tid] = 0.f;
}

// ---------------------------------------------------------------------------
// Kernel 1: fused scores (WMMA Gram) -> top-49 selection -> neighbor mean.
// One block = 4 waves handles a 16-row strip of the 1024x1024 score matrix,
// kept entirely in 64 KB of LDS as packed orderable (score, col) uint32 keys.
//   - sqrt dropped (monotonic), row norm dropped (constant per row)
//   - selection = 49 passes of pure u32-min trees + 5 shuffle-mins
// grid: (N/16, B), block: 128
// ---------------------------------------------------------------------------
__global__ __launch_bounds__(128) void knn_aggregate(
    const _Float16* __restrict__ Xh, const float* __restrict__ sq,
    _Float16* __restrict__ Zh) {
  __shared__ unsigned kdist[16 * N_DIM];  // 64 KB strip of packed keys

  const int lane = threadIdx.x & 31;
  const int wave = threadIdx.x >> 5;
  const int b = blockIdx.y;
  const int row_base = blockIdx.x * 16;
  const int m = lane & 15;
  const int hi = lane >> 4;

  const _Float16* xb = Xh + (size_t)b * N_DIM * C_DIM;
  const float* sqb = sq + b * N_DIM;

  // A fragments for this 16-row strip (K = 0..31 and 32..63), reused 16x.
  AFrag a0, a1;
  const _Float16* arow = xb + (size_t)(row_base + m) * C_DIM;
  a0.h[0] = *(const v8h*)(arow + 16 * hi);
  a0.h[1] = *(const v8h*)(arow + 8 + 16 * hi);
  a1.h[0] = *(const v8h*)(arow + 32 + 16 * hi);
  a1.h[1] = *(const v8h*)(arow + 40 + 16 * hi);

  // Each wave owns 16 column tiles (256 columns) of the strip.
  for (int ct = wave * 16; ct < wave * 16 + 16; ++ct) {
    const int col_base = ct * 16;
    const _Float16* brow = xb + (size_t)(col_base + m) * C_DIM;
    const v16h b0 = *(const v16h*)(brow + 16 * hi);
    const v16h b1 = *(const v16h*)(brow + 32 + 16 * hi);
    __builtin_prefetch(brow + 16 * C_DIM, 0, 1);  // next tile's row
    CFrag c = {};
    c.v = wmma_f16(a0.v, b0, c.v);
    c.v = wmma_f16(a1.v, b1, c.v);
    const float sqc = sqb[col_base + m];
    const int gc = col_base + m;
#pragma unroll
    for (int j = 0; j < 8; ++j) {
      const int gr = row_base + 8 * hi + j;
      // ranking key: sq_c - 2*G (row-constant sq_r and the sqrt are dropped;
      // both are monotonic w.r.t. the per-row ordering)
      const float score = fmaf(-2.f, c.f[j], sqc);
      const int ib = __float_as_int(score);
      unsigned u = (unsigned)(ib ^ ((ib >> 31) | 0x80000000));
      u = (u & 0xFFFFFC00u) | (unsigned)gc;   // embed column index
      if (gr == gc) u = 0xFFFFFFFFu;          // exclude self
      kdist[(8 * hi + j) * N_DIM + gc] = u;
    }
  }
  __syncthreads();

  // Selection + aggregation: one wave per row, 4 rows per wave.
  for (int rr = 0; rr < 4; ++rr) {
    const int lrow = wave * 4 + rr;
    const int grow = row_base + lrow;
    unsigned* drow = &kdist[lrow * N_DIM];
    float acc0 = 0.f, acc1 = 0.f;
    for (int s = 0; s < KNN; ++s) {
      // wave-wide min over 1024 packed keys (uint4 LDS reads, u32-min tree)
      unsigned bv = 0xFFFFFFFFu;
#pragma unroll
      for (int t = 0; t < 8; ++t) {
        const uint4 v = *(const uint4*)(drow + 128 * t + 4 * lane);
        bv = umin2(bv, umin2(umin2(v.x, v.y), umin2(v.z, v.w)));
      }
#pragma unroll
      for (int off = 16; off > 0; off >>= 1)
        bv = umin2(bv, (unsigned)__shfl_xor((int)bv, off, 32));
      const unsigned bidx = bv & 1023u;  // winner column falls out of the key
      drow[bidx] = 0xFFFFFFFFu;          // knockout (all lanes same addr/value)
      // gather neighbor features: each lane accumulates 2 channels
      const _Float16* nrow = xb + (size_t)bidx * C_DIM;
      const v2h hv = *(const v2h*)(nrow + 2 * lane);
      acc0 += (float)hv.x;
      acc1 += (float)hv.y;
    }
    _Float16* zrow = Zh + ((size_t)b * N_DIM + grow) * C_DIM;
    v2h zo;
    zo.x = (_Float16)(acc0 * (1.f / KNN));
    zo.y = (_Float16)(acc1 * (1.f / KNN));
    *(v2h*)(zrow + 2 * lane) = zo;
  }
}

// ---------------------------------------------------------------------------
// Kernel 2: Fg = BN(relu(W·Z + b)) then mean over nodes, via WMMA + fused
// epilogue + cross-lane reduction + atomicAdd into (B, O) output.
// grid: (O/16, B), block: 128 (4 waves, each covers 256 nodes)
// ---------------------------------------------------------------------------
__global__ __launch_bounds__(128) void gemm_bn_pool(
    const _Float16* __restrict__ Wh, const _Float16* __restrict__ Zh,
    const float* __restrict__ bias, const float* __restrict__ gamma,
    const float* __restrict__ beta, const float* __restrict__ rmean,
    const float* __restrict__ rvar, float* __restrict__ out) {
  const int lane = threadIdx.x & 31;
  const int wave = threadIdx.x >> 5;
  const int b = blockIdx.y;
  const int o_base = blockIdx.x * 16;
  const int m = lane & 15;
  const int hi = lane >> 4;

  AFrag a0, a1;
  const _Float16* arow = Wh + (size_t)(o_base + m) * C_DIM;
  a0.h[0] = *(const v8h*)(arow + 16 * hi);
  a0.h[1] = *(const v8h*)(arow + 8 + 16 * hi);
  a1.h[0] = *(const v8h*)(arow + 32 + 16 * hi);
  a1.h[1] = *(const v8h*)(arow + 40 + 16 * hi);

  // fold BN into scale/offset per output channel
  float pb[8], psc[8], pof[8];
#pragma unroll
  for (int j = 0; j < 8; ++j) {
    const int o = o_base + 8 * hi + j;
    pb[j] = bias[o];
    const float inv = rsqrtf(rvar[o] + BN_EPS);
    psc[j] = gamma[o] * inv;
    pof[j] = beta[o] - gamma[o] * rmean[o] * inv;
  }

  float sums[8] = {};
  const _Float16* zb = Zh + (size_t)b * N_DIM * C_DIM;
  for (int nt = wave * 16; nt < wave * 16 + 16; ++nt) {
    const int n_base = nt * 16;
    const _Float16* brow = zb + (size_t)(n_base + m) * C_DIM;
    const v16h b0 = *(const v16h*)(brow + 16 * hi);
    const v16h b1 = *(const v16h*)(brow + 32 + 16 * hi);
    __builtin_prefetch(brow + 16 * C_DIM, 0, 1);  // next tile's row
    CFrag c = {};
    c.v = wmma_f16(a0.v, b0, c.v);
    c.v = wmma_f16(a1.v, b1, c.v);
#pragma unroll
    for (int j = 0; j < 8; ++j) {
      const float v = fmaxf(c.f[j] + pb[j], 0.f);  // conv bias + ReLU
      sums[j] += v * psc[j] + pof[j];              // BatchNorm (eval)
    }
  }
  // reduce over the 16 N-columns held within each half-wave
#pragma unroll
  for (int off = 1; off < 16; off <<= 1) {
#pragma unroll
    for (int j = 0; j < 8; ++j) sums[j] += __shfl_xor(sums[j], off, 32);
  }
  if (m == 0) {
#pragma unroll
    for (int j = 0; j < 8; ++j)
      atomicAdd(&out[b * O_DIM + o_base + 8 * hi + j], sums[j] * (1.f / N_DIM));
  }
}

// ---------------------------------------------------------------------------
extern "C" void kernel_launch(void* const* d_in, const int* in_sizes, int n_in,
                              void* d_out, int out_size, void* d_ws, size_t ws_size,
                              hipStream_t stream) {
  (void)in_sizes; (void)n_in; (void)out_size; (void)ws_size;
  const float* X     = (const float*)d_in[0];
  const float* W     = (const float*)d_in[1];
  const float* bias  = (const float*)d_in[2];
  const float* gamma = (const float*)d_in[3];
  const float* beta  = (const float*)d_in[4];
  const float* rmean = (const float*)d_in[5];
  const float* rvar  = (const float*)d_in[6];
  float* out = (float*)d_out;

  char* ws = (char*)d_ws;
  _Float16* Xh = (_Float16*)(ws);                                 // 8 MB
  _Float16* Zh = (_Float16*)(ws + ((size_t)8 << 20));             // 8 MB
  _Float16* Wh = (_Float16*)(ws + ((size_t)16 << 20));            // 16 KB
  float*    sq = (float*)(ws + ((size_t)16 << 20) + (64u << 10)); // 256 KB

  hipLaunchKernelGGL(prep_nodes, dim3(256), dim3(256), 0, stream, X, Xh, sq);
  hipLaunchKernelGGL(prep_small, dim3(32), dim3(256), 0, stream, W, Wh, out);
  hipLaunchKernelGGL(knn_aggregate, dim3(N_DIM / 16, B_DIM), dim3(128), 0, stream,
                     Xh, sq, Zh);
  hipLaunchKernelGGL(gemm_bn_pool, dim3(O_DIM / 16, B_DIM), dim3(128), 0, stream,
                     Wh, Zh, bias, gamma, beta, rmean, rvar, out);
}
